// RNN_87986700025987
// MI455X (gfx1250) — compile-verified, matrix operand order checked by reading, not run on hardware
//
#include <hip/hip_runtime.h>
#include <stdint.h>
#include <math.h>

// ---------------------------------------------------------------------------
// MI455X (gfx1250) implementation.
//
// Only batch row 0 of each RNN reaches the output (zz = concat(e[0], o[0], z))
// and tanh-RNNs are batch-independent, so we evolve a single hidden row.
// The problem is scan-latency bound: 2 RNNs x 2 layers x 4096 sequential
// steps of a 64x64 GEMV + tanh.  Each GEMV is a chain of
// v_wmma_f32_16x16x32_f16 (4 N-tiles x 2 K-chunks) with the live row
// broadcast into all 16 A rows, so WMMA runs with full EXEC and every D row
// (and both lane halves) carries the live row -> no predication is needed
// anywhere in the hot loop (duplicate-lane LDS stores write identical data).
//
// The two layers + the layer-1 input projection are software-pipelined
// across 3 waves of one workgroup (one workgroup per RNN), one barrier per
// step; state ping-pongs through tiny LDS buffers re-read as aligned
// ds_load_b128 A-fragments.  Weight matrices are converted once into f16
// WMMA B-fragments resident in VGPRs for the whole scan.  Each step is
// phased WMMA-block -> VALU-block -> store-block so the WMMA->VALU hazard
// windows of early tiles are absorbed by later tiles' WMMAs.
// Total global traffic ~100 KB; HBM bandwidth is irrelevant.
// ---------------------------------------------------------------------------

typedef __attribute__((ext_vector_type(16))) _Float16 v16h;
typedef __attribute__((ext_vector_type(8)))  _Float16 v8h;
typedef __attribute__((ext_vector_type(8)))  float    v8f;

#define T_STEPS 4096
#define HID 64

__device__ __forceinline__ float fast_tanh(float x) {
#if __has_builtin(__builtin_amdgcn_tanhf)
  return __builtin_amdgcn_tanhf(x);   // gfx1250 v_tanh_f32 transcendental
#else
  return tanhf(x);
#endif
}

// Build one 32x16 f16 B-fragment for N-tile nt, K-chunk f, from row-major
// f32 W[64][64], realizing B[k][n] = W[n][k]  (i.e. h @ W^T).
__device__ __forceinline__ v16h make_bfrag(const float* W, int nt, int f, int lane) {
  const int n    = nt * 16 + (lane & 15);
  const int half = lane >> 4;
  v16h b;
#pragma unroll
  for (int v = 0; v < 8; ++v) {
    const int k = f * 32 + (v >> 2) * 16 + half * 8 + (v & 3) * 2;
    b[2 * v]     = (_Float16)W[n * 64 + k];
    b[2 * v + 1] = (_Float16)W[n * 64 + k + 1];
  }
  return b;
}

// Load a 16x32 f16 A-fragment for K-chunk f from a 64-entry f16 hidden-state
// vector in LDS.  Every row lane reads the single live row, so all 16 rows
// of A (and hence of D) are identical.  Two aligned ds_load_b128 per lane.
__device__ __forceinline__ v16h load_afrag(const _Float16* hbuf, int f, int lane) {
  const int half = lane >> 4;
  const int e0   = f * 32 + half * 8;            // elements for VGPRs 0..3
  v8h p0 = *(const v8h*)(hbuf + e0);             // K: e0 .. e0+7
  v8h p1 = *(const v8h*)(hbuf + e0 + 16);        // K: e0+16 .. e0+23
  return __builtin_shufflevector(p0, p1, 0, 1, 2, 3, 4, 5, 6, 7,
                                         8, 9, 10, 11, 12, 13, 14, 15);
}

// One 1x64 . 64x16 GEMV tile: two chained WMMAs over K=64.
__device__ __forceinline__ v8f wmma_k64(v16h a0, v16h a1, v16h b0, v16h b1) {
  v8f c = {};
  c = __builtin_amdgcn_wmma_f32_16x16x32_f16(false, a0, false, b0, (short)0, c, false, false);
  c = __builtin_amdgcn_wmma_f32_16x16x32_f16(false, a1, false, b1, (short)0, c, false, false);
  return c;
}

struct RnnParams {
  const float* x;      // [B, T, 3] -- only batch row 0 is live
  const float* Wih0;   // [64, 3]
  const float* Whh0;   // [64, 64]
  const float* bih0;   // [64]
  const float* bhh0;   // [64]
  const float* Wih1;   // [64, 64]
  const float* Whh1;   // [64, 64]
  const float* bih1;   // [64]
  const float* bhh1;   // [64]
};
struct ScanArgs { RnnParams r[2]; };

// One workgroup per RNN; 3 waves: 0=layer0, 1=layer1 input proj, 2=layer1.
__global__ __launch_bounds__(96) void rnn_scan_kernel(ScanArgs args, float* hfinal) {
  __shared__ alignas(16) _Float16 h0buf[2][HID];  // layer0 state, double buffered
  __shared__ alignas(16) _Float16 h1buf[HID];     // layer1 state (wave-2 private)
  __shared__ alignas(16) float    ubuf[2][HID];   // layer1 input projection

  const RnnParams P = args.r[blockIdx.x];
  const int tid  = threadIdx.x;
  const int w    = tid >> 5;       // wave role
  const int lane = tid & 31;
  const int ln   = lane & 15;

  if (tid < HID) {
    h0buf[0][tid] = (_Float16)0.f;
    h0buf[1][tid] = (_Float16)0.f;
    h1buf[tid]    = (_Float16)0.f;
  }

  // Per-role constants, resident in VGPRs for the whole scan.
  const float* Wsel = (w == 0) ? P.Whh0 : (w == 1) ? P.Wih1 : P.Whh1;
  v16h  bw[4][2];
  float wi0[4], wi1[4], wi2[4], bias[4];
#pragma unroll
  for (int nt = 0; nt < 4; ++nt) {
    bw[nt][0] = make_bfrag(Wsel, nt, 0, lane);
    bw[nt][1] = make_bfrag(Wsel, nt, 1, lane);
    const int n = nt * 16 + ln;
    if (w == 0) {
      wi0[nt]  = P.Wih0[n * 3 + 0];
      wi1[nt]  = P.Wih0[n * 3 + 1];
      wi2[nt]  = P.Wih0[n * 3 + 2];
      bias[nt] = P.bih0[n] + P.bhh0[n];
    } else if (w == 2) {
      bias[nt] = P.bih1[n] + P.bhh1[n];
    } else {
      wi0[nt] = wi1[nt] = wi2[nt] = bias[nt] = 0.f;
    }
  }
  __syncthreads();

  for (int i = 0; i < T_STEPS + 2; ++i) {
    if (w == 0 && i < T_STEPS) {
      // h0[i] = tanh(x_i . Wih0^T + b0 + h0[i-1] . Whh0^T)
      const float x0 = P.x[i * 3 + 0];       // uniform -> s_load, overlaps dscnt waits
      const float x1 = P.x[i * 3 + 1];
      const float x2 = P.x[i * 3 + 2];
      const v16h a0 = load_afrag(h0buf[(i + 1) & 1], 0, lane);
      const v16h a1 = load_afrag(h0buf[(i + 1) & 1], 1, lane);
      v8f c[4];
#pragma unroll
      for (int nt = 0; nt < 4; ++nt)
        c[nt] = wmma_k64(a0, a1, bw[nt][0], bw[nt][1]);
      float h[4];
#pragma unroll
      for (int nt = 0; nt < 4; ++nt) {
        const float pre = bias[nt] + x0 * wi0[nt] + x1 * wi1[nt] + x2 * wi2[nt];
        h[nt] = fast_tanh(c[nt][0] + pre);
      }
#pragma unroll
      for (int nt = 0; nt < 4; ++nt)            // duplicate-lane store, same data
        h0buf[i & 1][nt * 16 + ln] = (_Float16)h[nt];
    } else if (w == 1 && i >= 1 && i <= T_STEPS) {
      // u[t] = h0[t] . Wih1^T
      const int t = i - 1;
      const v16h a0 = load_afrag(h0buf[t & 1], 0, lane);
      const v16h a1 = load_afrag(h0buf[t & 1], 1, lane);
      v8f c[4];
#pragma unroll
      for (int nt = 0; nt < 4; ++nt)
        c[nt] = wmma_k64(a0, a1, bw[nt][0], bw[nt][1]);
#pragma unroll
      for (int nt = 0; nt < 4; ++nt)
        ubuf[t & 1][nt * 16 + ln] = c[nt][0];
    } else if (w == 2 && i >= 2) {
      // h1[t] = tanh(u[t] + b1 + h1[t-1] . Whh1^T)
      const int t = i - 2;
      const v16h a0 = load_afrag(h1buf, 0, lane);
      const v16h a1 = load_afrag(h1buf, 1, lane);
      v8f c[4];
#pragma unroll
      for (int nt = 0; nt < 4; ++nt)
        c[nt] = wmma_k64(a0, a1, bw[nt][0], bw[nt][1]);
      float h[4];
#pragma unroll
      for (int nt = 0; nt < 4; ++nt)
        h[nt] = fast_tanh(c[nt][0] + ubuf[t & 1][nt * 16 + ln] + bias[nt]);
#pragma unroll
      for (int nt = 0; nt < 4; ++nt)
        h1buf[nt * 16 + ln] = (_Float16)h[nt];
    }
    __syncthreads();
  }

  // Publish the final layer-1 hidden row (hoisted out of the hot loop).
  if (tid < HID) hfinal[blockIdx.x * HID + tid] = (float)h1buf[tid];
}

// Tiny head: e = hx.fce^T+b (row 0), o likewise, zz[9] -> fc1 -> relu -> fc2.
__global__ __launch_bounds__(64) void head_kernel(
    const float* hfinal, const float* z,
    const float* fce_w, const float* fce_b,
    const float* fco_w, const float* fco_b,
    const float* fc1_w, const float* fc1_b,
    const float* fc2_w, const float* fc2_b,
    float* out) {
  __shared__ float zz[9];
  __shared__ float hmid[64];
  const int tid = threadIdx.x;
  if (tid < 2) {
    float s = fce_b[tid];
    for (int k = 0; k < 64; ++k) s += hfinal[k] * fce_w[tid * 64 + k];
    zz[tid] = s;
  } else if (tid < 4) {
    const int q = tid - 2;
    float s = fco_b[q];
    for (int k = 0; k < 64; ++k) s += hfinal[64 + k] * fco_w[q * 64 + k];
    zz[tid] = s;
  } else if (tid < 9) {
    zz[tid] = z[tid - 4];
  }
  __syncthreads();
  {
    float s = fc1_b[tid];
    for (int k = 0; k < 9; ++k) s += zz[k] * fc1_w[tid * 9 + k];
    hmid[tid] = fmaxf(s, 0.f);
  }
  __syncthreads();
  if (tid < 5) {
    float s = fc2_b[tid];
    for (int k = 0; k < 64; ++k) s += hmid[k] * fc2_w[tid * 64 + k];
    out[tid] = s;
  }
}

extern "C" void kernel_launch(void* const* d_in, const int* in_sizes, int n_in,
                              void* d_out, int out_size, void* d_ws, size_t ws_size,
                              hipStream_t stream) {
  (void)in_sizes; (void)n_in; (void)out_size; (void)ws_size;
  auto F = [&](int i) { return (const float*)d_in[i]; };

  ScanArgs args;
  // rnn_empty on x
  args.r[0] = { F(0), F(3), F(4), F(5), F(6), F(7), F(8), F(9), F(10) };
  // rnn_occupied on y
  args.r[1] = { F(1), F(11), F(12), F(13), F(14), F(15), F(16), F(17), F(18) };

  float* hfinal = (float*)d_ws;   // [2][64] last hidden rows (batch row 0)

  rnn_scan_kernel<<<dim3(2), dim3(96), 0, stream>>>(args, hfinal);
  head_kernel<<<dim3(1), dim3(64), 0, stream>>>(
      hfinal, F(2), F(19), F(20), F(21), F(22),
      F(23), F(24), F(25), F(26), (float*)d_out);
}
